// MambaVision_Mixer_22909355557350
// MI455X (gfx1250) — matrix operand order repaired and checked by reading, not compiled, not run on hardware
//
#include <hip/hip_runtime.h>
#include <hip/hip_bf16.h>
#include <math.h>

// ---------------------------------------------------------------------------
// MambaVision mixer for MI455X (gfx1250, wave32, WMMA + async LDS staging).
// B=8, L=2048, D_MODEL=512, D_INNER=512, HALF=256, D_STATE=32, DT_RANK=32.
// ---------------------------------------------------------------------------

#define B_SZ    8
#define L_SZ    2048
#define DMODEL  512
#define DINNER  512
#define HALF_C  256
#define NSTATE  32
#define DTRANK  32
#define XDBL_C  96           // dt_rank + 2*d_state
#define MTOT    (B_SZ * L_SZ)

typedef __attribute__((ext_vector_type(16))) __bf16 v16bf;
typedef __attribute__((ext_vector_type(8)))  __bf16 v8bf;
typedef __attribute__((ext_vector_type(4)))  __bf16 v4bf;
typedef __attribute__((ext_vector_type(8)))  float  v8f;
typedef int v4i __attribute__((vector_size(4 * sizeof(int))));  // int4 vector

#if defined(__has_builtin)
#  if __has_builtin(__builtin_amdgcn_global_load_async_to_lds_b128)
#    define ASYNC_LDS 1
#  endif
#endif
#ifndef ASYNC_LDS
#  define ASYNC_LDS 0
#endif

__device__ __forceinline__ void async_wait0() {
#if ASYNC_LDS
#  if defined(__has_builtin) && __has_builtin(__builtin_amdgcn_s_wait_asynccnt)
    __builtin_amdgcn_s_wait_asynccnt(0);
#  else
    asm volatile("s_wait_asynccnt 0x0" ::: "memory");
#  endif
#endif
}

// --- fragment loaders -------------------------------------------------------

// bf16 source (global or LDS): two 16-byte loads, concatenated.
__device__ __forceinline__ v16bf load_frag_bf(const __bf16* p0, const __bf16* p1) {
    v8bf a = *reinterpret_cast<const v8bf*>(p0);
    v8bf b = *reinterpret_cast<const v8bf*>(p1);
    return __builtin_shufflevector(a, b, 0, 1, 2, 3, 4, 5, 6, 7,
                                         8, 9, 10, 11, 12, 13, 14, 15);
}

// f32 source: 8+8 floats, converted.
__device__ __forceinline__ v16bf load_frag_f32(const float* p0, const float* p1) {
    const float4* q0 = reinterpret_cast<const float4*>(p0);
    const float4* q1 = reinterpret_cast<const float4*>(p1);
    float4 f0 = q0[0], f1 = q0[1];
    float4 f2 = q1[0], f3 = q1[1];
    v16bf r;
    r[0]  = (__bf16)f0.x; r[1]  = (__bf16)f0.y; r[2]  = (__bf16)f0.z; r[3]  = (__bf16)f0.w;
    r[4]  = (__bf16)f1.x; r[5]  = (__bf16)f1.y; r[6]  = (__bf16)f1.z; r[7]  = (__bf16)f1.w;
    r[8]  = (__bf16)f2.x; r[9]  = (__bf16)f2.y; r[10] = (__bf16)f2.z; r[11] = (__bf16)f2.w;
    r[12] = (__bf16)f3.x; r[13] = (__bf16)f3.y; r[14] = (__bf16)f3.z; r[15] = (__bf16)f3.w;
    return r;
}

// --- WMMA GEMM --------------------------------------------------------------
// C[m,n] = sum_k A[m,k] * Bw[n,k]   (Bw row-major N x K, i.e. "x @ W^T").
// Each wave computes a (16*MT) x (16*NT) tile; 4 waves/block stacked in M.
// The shared B tile (16*NT cols x 32 k) is staged into LDS with async
// global->LDS copies, double-buffered across the K loop.
// EPI: 0 = +bias, 1 = raw, 2 = +bias then softplus.
template <int EPI, bool ABF, int MT, int NT>
__global__ __launch_bounds__(128) void wmma_gemm_kernel(
    const void* __restrict__ Ap, int lda,
    const __bf16* __restrict__ Bw, int ldb,
    const float* __restrict__ bias,
    float* __restrict__ C, int ldc,
    int K) {
    constexpr int BPITCH = 40;                    // 80B rows: bank-spread, 16B aligned
    __shared__ __bf16 Bsh[2][16 * NT][BPITCH];

    const int tid    = threadIdx.x;
    const int lane   = tid & 31;
    const int warp   = tid >> 5;
    const int half   = lane >> 4;                 // 0: lanes 0-15, 1: lanes 16-31
    const int l15    = lane & 15;
    const int m_base = (blockIdx.x * 4 + warp) * (16 * MT);
    const int n_base = blockIdx.y * (16 * NT);

    const __bf16* Abf = (const __bf16*)Ap;
    const float*  Af  = (const float*)Ap;

    // Cooperative stage of one 16*NT x 32 bf16 B tile into Bsh[buf].
    auto stage_B = [&](int buf, int k0) {
        for (int idx = tid; idx < NT * 16 * 4; idx += 128) {
            const int col = idx >> 2, chunk = idx & 3;
            const __bf16* src = Bw + (size_t)(n_base + col) * ldb + k0 + chunk * 8;
            __bf16*       dst = &Bsh[buf][col][chunk * 8];
#if ASYNC_LDS
            __builtin_amdgcn_global_load_async_to_lds_b128(
                (__attribute__((address_space(1))) v4i*)src,
                (__attribute__((address_space(3))) v4i*)dst, 0, 0);
#else
            *reinterpret_cast<v8bf*>(dst) = *reinterpret_cast<const v8bf*>(src);
#endif
        }
    };

    v8f acc[MT][NT];
#pragma unroll
    for (int mt = 0; mt < MT; ++mt)
#pragma unroll
        for (int nt = 0; nt < NT; ++nt)
            acc[mt][nt] = v8f{0.f, 0.f, 0.f, 0.f, 0.f, 0.f, 0.f, 0.f};

    stage_B(0, 0);
    async_wait0();
    __syncthreads();

    int buf = 0;
    for (int k0 = 0; k0 < K; k0 += 32) {
        // Kick off the next B tile while this one computes.
        if (k0 + 32 < K) stage_B(buf ^ 1, k0 + 32);

        // A fragments (16x32): lanes 0-15 -> row M=lane, K {0..7,16..23};
        // lanes 16-31 -> row M=lane-16, K {8..15,24..31}.  Direct global loads.
        v16bf afrag[MT];
#pragma unroll
        for (int mt = 0; mt < MT; ++mt) {
            const int row = m_base + mt * 16 + l15;
            if (ABF) {
                const __bf16* r0 = Abf + (size_t)row * lda + k0 + (half ? 8 : 0);
                const __bf16* r1 = Abf + (size_t)row * lda + k0 + (half ? 24 : 16);
                afrag[mt] = load_frag_bf(r0, r1);
            } else {
                const float* r0 = Af + (size_t)row * lda + k0 + (half ? 8 : 0);
                const float* r1 = Af + (size_t)row * lda + k0 + (half ? 24 : 16);
                afrag[mt] = load_frag_f32(r0, r1);
            }
        }
        if (k0 + 32 < K) {
            const void* pa = ABF
                ? (const void*)(Abf + (size_t)(m_base + l15) * lda + k0 + 32)
                : (const void*)(Af + (size_t)(m_base + l15) * lda + k0 + 32);
            __builtin_prefetch(pa, 0, 3);
        }

        // B fragments (32x16) from LDS: lane = column n; lanes 0-15 hold
        // K {0..7,16..23}, lanes 16-31 hold K {8..15,24..31}.
        v16bf bfrag[NT];
#pragma unroll
        for (int nt = 0; nt < NT; ++nt) {
            const __bf16* p = &Bsh[buf][nt * 16 + l15][half * 8];
            bfrag[nt] = load_frag_bf(p, p + 16);
        }

#pragma unroll
        for (int mt = 0; mt < MT; ++mt)
#pragma unroll
            for (int nt = 0; nt < NT; ++nt)
                acc[mt][nt] = __builtin_amdgcn_wmma_f32_16x16x32_bf16(
                    false, afrag[mt], false, bfrag[nt],
                    (short)0, acc[mt][nt], false, false);

        // Retire the in-flight stage and flip buffers.
        async_wait0();
        __syncthreads();
        buf ^= 1;
    }

    // C/D layout: VGPR r -> M = r + 8*half, N = lane&15.
#pragma unroll
    for (int nt = 0; nt < NT; ++nt) {
        const int col = n_base + nt * 16 + l15;
        float bv = 0.f;
        if (EPI != 1) bv = bias[col];
#pragma unroll
        for (int mt = 0; mt < MT; ++mt) {
#pragma unroll
            for (int r = 0; r < 8; ++r) {
                const int row = m_base + mt * 16 + r + half * 8;
                float v = acc[mt][nt][r];
                if (EPI != 1) v += bv;
                if (EPI == 2) v = (v > 20.f) ? v : log1pf(__expf(v));  // softplus
                C[(size_t)row * ldc + col] = v;
            }
        }
    }
}

// --- f32 -> bf16 conversion (vectorized x4) ---------------------------------
__global__ __launch_bounds__(256) void f32_to_bf16_kernel(
    const float* __restrict__ in, __bf16* __restrict__ out, int n4) {
    const int i = blockIdx.x * blockDim.x + threadIdx.x;
    if (i < n4) {
        float4 f = reinterpret_cast<const float4*>(in)[i];
        v4bf o;
        o[0] = (__bf16)f.x; o[1] = (__bf16)f.y;
        o[2] = (__bf16)f.z; o[3] = (__bf16)f.w;
        reinterpret_cast<v4bf*>(out)[i] = o;
    }
}

// --- depthwise conv (k=4, pad 1/2) + SiLU -----------------------------------
__global__ __launch_bounds__(256) void conv_silu_kernel(
    const float* __restrict__ xz,       // (b,l,512)
    const float* __restrict__ wx,       // (256,1,4)
    const float* __restrict__ wz,       // (256,1,4)
    float* __restrict__ x_act,          // (b,l,256) f32 (scan input u)
    __bf16* __restrict__ x_act_bf,      // (b,l,256) bf16 (x_proj GEMM input)
    __bf16* __restrict__ ycat_bf) {     // (b,l,512) bf16: z half -> [256:512]
    const int c  = threadIdx.x;         // 0..255
    const int b  = blockIdx.y;
    const int l0 = blockIdx.x * 64;

    float wxc[4], wzc[4];
#pragma unroll
    for (int j = 0; j < 4; ++j) { wxc[j] = wx[c * 4 + j]; wzc[j] = wz[c * 4 + j]; }

    for (int i = 0; i < 64; ++i) {
        const int l = l0 + i;
        float ax = 0.f, az = 0.f;
#pragma unroll
        for (int j = 0; j < 4; ++j) {
            const int ll = l + j - 1;
            if (ll >= 0 && ll < L_SZ) {
                const float* p = xz + ((size_t)(b * L_SZ + ll)) * DINNER;
                ax += p[c] * wxc[j];
                az += p[HALF_C + c] * wzc[j];
            }
        }
        ax = ax / (1.f + __expf(-ax));   // SiLU
        az = az / (1.f + __expf(-az));
        const size_t m = (size_t)(b * L_SZ + l);
        x_act   [m * HALF_C + c] = ax;
        x_act_bf[m * HALF_C + c] = (__bf16)ax;
        ycat_bf [m * DINNER + HALF_C + c] = (__bf16)az;
    }
}

// --- selective scan: one wave32 per (b,d), lane = state index n -------------
__global__ __launch_bounds__(256) void scan_kernel(
    const float* __restrict__ x_act,   // u,     (b,l,256)
    const float* __restrict__ delta,   //        (b,l,256)
    const float* __restrict__ xdbl,    //        (b,l,96): [32:64]=B, [64:96]=C
    const float* __restrict__ A_log,   // (256,32)
    const float* __restrict__ Dvec,    // (256)
    __bf16* __restrict__ ycat_bf) {    // (b,l,512): y half -> [0:256]
    const int lane = threadIdx.x & 31;
    const int wg   = blockIdx.x * 8 + (threadIdx.x >> 5);   // 0..2047
    const int b    = wg >> 8;
    const int d    = wg & 255;

    const float Ad = -__expf(A_log[d * NSTATE + lane]);
    const float Dd = Dvec[d];

    const float* xa = x_act + (size_t)b * L_SZ * HALF_C + d;
    const float* dl = delta + (size_t)b * L_SZ * HALF_C + d;
    const float* xb = xdbl  + (size_t)b * L_SZ * XDBL_C;
    __bf16*      yo = ycat_bf + (size_t)b * L_SZ * DINNER + d;

    float state = 0.f;
    for (int l = 0; l < L_SZ; ++l) {
        const float dt = dl[(size_t)l * HALF_C];
        const float u  = xa[(size_t)l * HALF_C];
        const float Bn = xb[(size_t)l * XDBL_C + DTRANK + lane];
        const float Cn = xb[(size_t)l * XDBL_C + DTRANK + NSTATE + lane];

        state = state * __expf(dt * Ad) + (dt * u) * Bn;

        float p = state * Cn;
#pragma unroll
        for (int off = 16; off >= 1; off >>= 1) p += __shfl_xor(p, off, 32);
        if (lane == 0) yo[(size_t)l * DINNER] = (__bf16)(p + Dd * u);
    }
}

// ---------------------------------------------------------------------------

extern "C" void kernel_launch(void* const* d_in, const int* in_sizes, int n_in,
                              void* d_out, int out_size, void* d_ws, size_t ws_size,
                              hipStream_t stream) {
    const float* hidden = (const float*)d_in[0];   // (8,2048,512)
    const float* in_w   = (const float*)d_in[1];   // (512,512)
    const float* in_b   = (const float*)d_in[2];   // (512)
    const float* convx  = (const float*)d_in[3];   // (256,1,4)
    const float* convz  = (const float*)d_in[4];   // (256,1,4)
    const float* xprojw = (const float*)d_in[5];   // (96,256)
    const float* dtw    = (const float*)d_in[6];   // (256,32)
    const float* dtb    = (const float*)d_in[7];   // (256)
    const float* alog   = (const float*)d_in[8];   // (256,32)
    const float* dvec   = (const float*)d_in[9];   // (256)
    const float* outw   = (const float*)d_in[10];  // (512,512)
    const float* outb   = (const float*)d_in[11];  // (512)
    float*       out    = (float*)d_out;           // (8,2048,512) f32

    // ---- workspace layout: f32 region then bf16 region ----
    float* wf = (float*)d_ws;
    float* xz    = wf;                                   // (b,l,512) f32
    float* x_act = xz    + (size_t)MTOT * DINNER;        // (b,l,256) f32
    float* xdbl  = x_act + (size_t)MTOT * HALF_C;        // (b,l,96)  f32
    float* delta = xdbl  + (size_t)MTOT * XDBL_C;        // (b,l,256) f32

    __bf16* hidden_bf = (__bf16*)(delta + (size_t)MTOT * HALF_C);   // (b,l,512)
    __bf16* x_act_bf  = hidden_bf + (size_t)MTOT * DMODEL;   // (b,l,256)
    __bf16* ycat_bf   = x_act_bf  + (size_t)MTOT * HALF_C;   // (b,l,512)
    __bf16* w_in_bf   = ycat_bf   + (size_t)MTOT * DINNER;   // (512,512)
    __bf16* w_xp_bf   = w_in_bf   + (size_t)DINNER * DMODEL; // (96,256)
    __bf16* w_dt_bf   = w_xp_bf   + (size_t)XDBL_C * HALF_C; // (256,32)
    __bf16* w_out_bf  = w_dt_bf   + (size_t)HALF_C * DTRANK; // (512,512)

    // ---- 0) one-time f32 -> bf16 conversions ----
    auto cvt = [&](const float* src, __bf16* dst, int n) {
        const int n4 = n / 4;
        f32_to_bf16_kernel<<<(n4 + 255) / 256, 256, 0, stream>>>(src, dst, n4);
    };
    cvt(hidden, hidden_bf, MTOT * DMODEL);
    cvt(in_w,   w_in_bf,   DINNER * DMODEL);
    cvt(xprojw, w_xp_bf,   XDBL_C * HALF_C);
    cvt(dtw,    w_dt_bf,   HALF_C * DTRANK);
    cvt(outw,   w_out_bf,  DINNER * DMODEL);

    // ---- 1) xz = hidden @ in_proj_w^T + b     (16384 x 512 x 512) ----
    wmma_gemm_kernel<0, true, 2, 4><<<dim3(MTOT / 128, DINNER / 64), 128, 0, stream>>>(
        hidden_bf, DMODEL, w_in_bf, DMODEL, in_b, xz, DINNER, DMODEL);

    // ---- 2) depthwise conv + SiLU -> x_act{,_bf}, ycat_bf[z half] ----
    conv_silu_kernel<<<dim3(L_SZ / 64, B_SZ), 256, 0, stream>>>(
        xz, convx, convz, x_act, x_act_bf, ycat_bf);

    // ---- 3) xdbl = x_act @ x_proj_w^T         (16384 x 96 x 256) ----
    wmma_gemm_kernel<1, true, 2, 3><<<dim3(MTOT / 128, XDBL_C / 48), 128, 0, stream>>>(
        x_act_bf, HALF_C, w_xp_bf, HALF_C, nullptr, xdbl, XDBL_C, HALF_C);

    // ---- 4) delta = softplus(xdbl[:, :32] @ dt_proj_w^T + b)  (16384 x 256 x 32) ----
    wmma_gemm_kernel<2, false, 2, 4><<<dim3(MTOT / 128, HALF_C / 64), 128, 0, stream>>>(
        xdbl, XDBL_C, w_dt_bf, DTRANK, dtb, delta, HALF_C, DTRANK);

    // ---- 5) selective scan -> ycat_bf[y half] ----
    scan_kernel<<<dim3(2048 / 8), 256, 0, stream>>>(
        x_act, delta, xdbl, alog, dvec, ycat_bf);

    // ---- 6) out = ycat @ out_proj_w^T + b     (16384 x 512 x 512) ----
    wmma_gemm_kernel<0, true, 2, 4><<<dim3(MTOT / 128, DMODEL / 64), 128, 0, stream>>>(
        ycat_bf, DINNER, w_out_bf, DINNER, outb, out, DMODEL, DINNER);
}
